// Skipgram_Net_10754598109706
// MI455X (gfx1250) — matrix operand order repaired and checked by reading, not compiled, not run on hardware
//
#include <hip/hip_runtime.h>
#include <hip/hip_bf16.h>
#include <math.h>

// Problem constants (fixed by the reference).
#define VOCAB 100000
#define EMB   128
#define BATCH 2048
#define NTILE (VOCAB / 16)          // 6250 column tiles of 16
#define TPW   8                     // column tiles per wave
#define WAVES 8                     // waves per workgroup (256 threads, wave32)
#define TILES_PER_WG (WAVES * TPW)  // 64 tiles = 1024 columns per WG
#define NCB  ((NTILE + TILES_PER_WG - 1) / TILES_PER_WG)  // 98 column blocks
#define NROWT (BATCH / 16)          // 128 row tiles
#define NTILE_PAD (NCB * TILES_PER_WG)   // 6272 tiles (uniform trip counts)
#define VPAD  (NTILE_PAD * 16)           // 100352 padded vocab rows

// sched_barrier mask: VALU | SALU | TRANS may cross; VMEM + WMMA pinned.
#define SCHED_PIN_MEM_MMA (0x2 | 0x4 | 0x400)

typedef __attribute__((ext_vector_type(16))) __bf16 v16bf;
typedef __attribute__((ext_vector_type(8)))  float  v8f;
typedef __attribute__((ext_vector_type(4)))  float  f4;

union BfFrag { f4 q[2]; v16bf v; };
struct BTile { BfFrag b[4]; };   // 16 cols x 128 K of bf16 (32 VGPRs)

__device__ __forceinline__ unsigned short f2bf(float f) {
  unsigned int u = __float_as_uint(f);
  u += 0x7FFFu + ((u >> 16) & 1u);   // round to nearest even
  return (unsigned short)(u >> 16);
}

// ---------------- prep kernels ----------------

// Convert W_out [V,E] f32 -> bf16, zero-fill the pad rows [VOCAB, VPAD).
__global__ __launch_bounds__(256) void k_conv_wout(const float* __restrict__ W,
                                                   unsigned short* __restrict__ Wb) {
  int i = (blockIdx.x * 256 + threadIdx.x) * 4;    // VPAD*EMB/4 threads
  ushort4 o;
  if (i < VOCAB * EMB) {
    f4 v = *(const f4*)(W + i);
    o.x = f2bf(v.x); o.y = f2bf(v.y); o.z = f2bf(v.z); o.w = f2bf(v.w);
  } else {
    o.x = 0; o.y = 0; o.z = 0; o.w = 0;
  }
  *(ushort4*)(Wb + i) = o;
}

// Padded bias: pad entries are -inf so pad columns add exp(-inf)=0 to stats.
__global__ __launch_bounds__(256) void k_bias(const float* __restrict__ bo,
                                              float* __restrict__ bp) {
  int i = blockIdx.x * 256 + threadIdx.x;          // VPAD threads
  bp[i] = (i < VOCAB) ? bo[i] : -INFINITY;
}

// emb[b,e] = W_center[e, cw[b]] + b_center[e], stored bf16 row-major [B,E].
__global__ __launch_bounds__(256) void k_emb(const int* __restrict__ cw,
                                             const float* __restrict__ Wc,
                                             const float* __restrict__ bc,
                                             unsigned short* __restrict__ Eb) {
  int t = blockIdx.x * 256 + threadIdx.x;          // BATCH*EMB = 262144
  int b = t >> 7, e = t & 127;
  int idx = cw[b];
  Eb[t] = f2bf(Wc[(size_t)e * VOCAB + idx] + bc[e]);
}

// ---------------- shared GEMM helpers ----------------

__device__ __forceinline__ void load_afrag(BfFrag* a, const unsigned short* Eb,
                                           int rowtile, int l16, int hi) {
  // 16-bit A layout: lane holds row M=l16; K = kc*32 + hi*8 + {0..7} and +16.
  const unsigned short* eb = Eb + (size_t)(rowtile * 16 + l16) * EMB + hi * 8;
#pragma unroll
  for (int kc = 0; kc < 4; kc++) {
    a[kc].q[0] = *(const f4*)(eb + kc * 32);
    a[kc].q[1] = *(const f4*)(eb + kc * 32 + 16);
  }
}

// Issue all 8 b128 loads for one 16x16 tile's B data (no waits between them).
__device__ __forceinline__ void load_btile(BTile& t, const unsigned short* wb) {
#pragma unroll
  for (int kc = 0; kc < 4; kc++) {
    t.b[kc].q[0] = *(const f4*)(wb + kc * 32);
    t.b[kc].q[1] = *(const f4*)(wb + kc * 32 + 16);
  }
}

// 4 back-to-back WMMAs over K=128 using preloaded fragments.
__device__ __forceinline__ v8f wmma4(const BfFrag* a, const BTile& t) {
  v8f c = {};
#pragma unroll
  for (int kc = 0; kc < 4; kc++) {
    c = __builtin_amdgcn_wmma_f32_16x16x32_bf16(false, a[kc].v, false, t.b[kc].v,
                                                (short)0, c, false, false);
  }
  return c;
}

// ---------------- pass 1: GEMM + online softmax stats ----------------

__global__ __launch_bounds__(256) void k_gemm_stats(const unsigned short* __restrict__ Wb,
                                                    const unsigned short* __restrict__ Eb,
                                                    const float* __restrict__ biasP,
                                                    float* __restrict__ stats) {
  __shared__ float lm[WAVES][16], ls[WAVES][16];
  const int lane = threadIdx.x & 31;
  const int wave = threadIdx.x >> 5;
  const int l16  = lane & 15;
  const int hi   = lane >> 4;
  const int rowtile = blockIdx.x;

  BfFrag a[4];
  load_afrag(a, Eb, rowtile, l16, hi);

  float rm[8], rs[8];
#pragma unroll
  for (int r = 0; r < 8; r++) { rm[r] = -1e30f; rs[r] = 0.f; }

  const int ct0 = blockIdx.y * TILES_PER_WG + wave * TPW;      // wave-uniform
  const unsigned short* wb = Wb + ((size_t)ct0 * 16 + l16) * EMB + hi * 8;

  BTile t[2];
  float bb[2];
  load_btile(t[0], wb);
  bb[0] = biasP[ct0 * 16 + l16];

#pragma unroll
  for (int i = 0; i < TPW; i++) {
    const int cur = i & 1, nx = cur ^ 1;
    if (i + 1 < TPW) {                       // prefetch next tile (uniform)
      load_btile(t[nx], wb + (size_t)(i + 1) * 16 * EMB);
      bb[nx] = biasP[(ct0 + i + 1) * 16 + l16];
    }
    __builtin_amdgcn_sched_barrier(SCHED_PIN_MEM_MMA);
    v8f c = wmma4(a, t[cur]);                // waits only for tile i's loads
    __builtin_amdgcn_sched_barrier(SCHED_PIN_MEM_MMA);
    float bo = bb[cur];
#pragma unroll
    for (int r = 0; r < 8; r++) {
      float v  = c[r] + bo;
      float mn = fmaxf(rm[r], v);
      rs[r] = rs[r] * __expf(rm[r] - mn) + __expf(v - mn);
      rm[r] = mn;
    }
  }

  // merge across the 16 lanes that share each row (xor<16 stays in half-wave)
#pragma unroll
  for (int r = 0; r < 8; r++) {
#pragma unroll
    for (int d = 1; d < 16; d <<= 1) {
      float om = __shfl_xor(rm[r], d);
      float os = __shfl_xor(rs[r], d);
      float mn = fmaxf(rm[r], om);
      rs[r] = rs[r] * __expf(rm[r] - mn) + os * __expf(om - mn);
      rm[r] = mn;
    }
  }
  if (l16 == 0) {          // lanes 0 (rows 0..7) and 16 (rows 8..15)
#pragma unroll
    for (int r = 0; r < 8; r++) {
      lm[wave][hi * 8 + r] = rm[r];
      ls[wave][hi * 8 + r] = rs[r];
    }
  }
  __syncthreads();
  if (threadIdx.x < 16) {  // merge the 8 waves
    float m = -1e30f, s = 0.f;
#pragma unroll
    for (int w = 0; w < WAVES; w++) {
      float mb = lm[w][threadIdx.x], sb = ls[w][threadIdx.x];
      float mn = fmaxf(m, mb);
      s = s * __expf(m - mn) + sb * __expf(mb - mn);
      m = mn;
    }
    size_t idx = (((size_t)rowtile * NCB + blockIdx.y) * 16 + threadIdx.x) * 2;
    stats[idx]     = m;
    stats[idx + 1] = s;
  }
}

// ---------------- pass 2: per-row offset = m + ln(sumexp) ----------------

__global__ __launch_bounds__(256) void k_combine(const float* __restrict__ stats,
                                                 float* __restrict__ offs) {
  int b = blockIdx.x * 256 + threadIdx.x;
  if (b >= BATCH) return;
  int rt = b >> 4, ri = b & 15;
  float m = -1e30f, s = 0.f;
  for (int cb = 0; cb < NCB; cb++) {
    size_t idx = (((size_t)rt * NCB + cb) * 16 + ri) * 2;
    float mb = stats[idx], sb = stats[idx + 1];
    float mn = fmaxf(m, mb);
    s = s * __expf(m - mn) + sb * __expf(mb - mn);
    m = mn;
  }
  offs[b] = m + __logf(s);
}

// ---------------- pass 3: recompute GEMM (B in L2), write log_softmax ----------------

__global__ __launch_bounds__(256) void k_gemm_write(const unsigned short* __restrict__ Wb,
                                                    const unsigned short* __restrict__ Eb,
                                                    const float* __restrict__ biasP,
                                                    const float* __restrict__ offs,
                                                    float* __restrict__ out) {
  const int lane = threadIdx.x & 31;
  const int wave = threadIdx.x >> 5;
  const int l16  = lane & 15;
  const int hi   = lane >> 4;
  const int rowtile = blockIdx.x;

  BfFrag a[4];
  load_afrag(a, Eb, rowtile, l16, hi);

  float off[8];
#pragma unroll
  for (int r = 0; r < 8; r++) off[r] = offs[rowtile * 16 + hi * 8 + r];

  const int ct0 = blockIdx.y * TILES_PER_WG + wave * TPW;
  const unsigned short* wb = Wb + ((size_t)ct0 * 16 + l16) * EMB + hi * 8;

  BTile t[2];
  float bb[2];
  load_btile(t[0], wb);
  bb[0] = biasP[ct0 * 16 + l16];

#pragma unroll
  for (int i = 0; i < TPW; i++) {
    const int cur = i & 1, nx = cur ^ 1;
    if (i + 1 < TPW) {
      load_btile(t[nx], wb + (size_t)(i + 1) * 16 * EMB);
      bb[nx] = biasP[(ct0 + i + 1) * 16 + l16];
    }
    __builtin_amdgcn_sched_barrier(SCHED_PIN_MEM_MMA);
    v8f c = wmma4(a, t[cur]);          // compute even for pad tiles (EXEC full)
    __builtin_amdgcn_sched_barrier(SCHED_PIN_MEM_MMA);
    int ct = ct0 + i;
    if (ct < NTILE) {                  // wave-uniform store guard
      float bo = bb[cur];
      int col = ct * 16 + l16;
      float* o = out + (size_t)(rowtile * 16 + hi * 8) * VOCAB + col;
#pragma unroll
      for (int r = 0; r < 8; r++) {
        o[(size_t)r * VOCAB] = c[r] + bo - off[r];   // VGPR r -> row r(+8)
      }
    }
  }
}

// ---------------- launch ----------------

extern "C" void kernel_launch(void* const* d_in, const int* in_sizes, int n_in,
                              void* d_out, int out_size, void* d_ws, size_t ws_size,
                              hipStream_t stream) {
  (void)in_sizes; (void)n_in; (void)out_size; (void)ws_size;
  const int*   cw = (const int*)d_in[0];
  const float* Wc = (const float*)d_in[1];
  const float* bc = (const float*)d_in[2];
  const float* Wo = (const float*)d_in[3];
  const float* bo = (const float*)d_in[4];
  float* out = (float*)d_out;

  // Workspace layout (256B-aligned offsets):
  const size_t off_wb    = 0;                                       // VPAD*EMB*2
  const size_t off_eb    = off_wb   + (size_t)VPAD * EMB * 2;       // BATCH*EMB*2
  const size_t off_bias  = off_eb   + (size_t)BATCH * EMB * 2;      // VPAD*4
  const size_t off_stats = off_bias + (size_t)VPAD * 4;             // 128*98*16*2*4
  const size_t off_offs  = off_stats + (size_t)NROWT * NCB * 16 * 2 * sizeof(float);

  char* ws = (char*)d_ws;
  unsigned short* Wb    = (unsigned short*)(ws + off_wb);
  unsigned short* Eb    = (unsigned short*)(ws + off_eb);
  float*          biasP = (float*)(ws + off_bias);
  float*          stats = (float*)(ws + off_stats);
  float*          offs  = (float*)(ws + off_offs);

  k_conv_wout<<<(VPAD * EMB) / 4 / 256, 256, 0, stream>>>(Wo, Wb);
  k_bias<<<VPAD / 256, 256, 0, stream>>>(bo, biasP);
  k_emb<<<(BATCH * EMB) / 256, 256, 0, stream>>>(cw, Wc, bc, Eb);

  dim3 grid(NROWT, NCB);
  k_gemm_stats<<<grid, 256, 0, stream>>>(Wb, Eb, biasP, stats);
  k_combine<<<(BATCH + 255) / 256, 256, 0, stream>>>(stats, offs);
  k_gemm_write<<<grid, 256, 0, stream>>>(Wb, Eb, biasP, offs, out);
}
